// LearntNeighbourhoodSampling_29703993819760
// MI455X (gfx1250) — matrix operand order repaired and checked by reading, not compiled, not run on hardware
//
#include <hip/hip_runtime.h>

// ---------------------------------------------------------------------------
// LearntNeighbourhoodSampling for MI455X (gfx1250, wave32, WMMA bf16 + TDM)
// B=2, N=32768, C=64, vol=64^3.  Whole linear chain folded into 27 effective
// [64x67] matrices -> 27x K=64 GEMMs done with v_wmma_f32_16x16x32_bf16.
// Folded weights (216 KB) are DMA'd into LDS once via TENSOR_LOAD_TO_LDS.
// ---------------------------------------------------------------------------

#define DHW   262144   // 64*64*64
#define NVERT 32768
#define NPTS  65536    // B*N
#define NB    2

typedef __bf16 bf16;
typedef bf16  v16bf __attribute__((ext_vector_type(16)));
typedef float v8f   __attribute__((ext_vector_type(8)));
typedef unsigned int u32x4 __attribute__((ext_vector_type(4)));
typedef int i32x4 __attribute__((ext_vector_type(4)));
typedef int i32x8 __attribute__((ext_vector_type(8)));

__device__ inline unsigned short f2bf(float f) {
  unsigned int u = __float_as_uint(f);
  u += 0x7FFFu + ((u >> 16) & 1u);          // round-to-nearest-even
  return (unsigned short)(u >> 16);
}

struct Tri {
  int i000, i001, i010, i011, i100, i101, i110, i111;
  float wx, wy, wz;
};

__device__ inline Tri tri_setup(float cx, float cy, float cz) {
  Tri t;
  float x = fminf(fmaxf((cx + 1.f) * 31.5f, 0.f), 63.f);
  float y = fminf(fmaxf((cy + 1.f) * 31.5f, 0.f), 63.f);
  float z = fminf(fmaxf((cz + 1.f) * 31.5f, 0.f), 63.f);
  int x0 = (int)floorf(x), y0 = (int)floorf(y), z0 = (int)floorf(z);
  int x1 = (x0 < 63) ? x0 + 1 : 63;
  int y1 = (y0 < 63) ? y0 + 1 : 63;
  int z1 = (z0 < 63) ? z0 + 1 : 63;
  t.wx = x - (float)x0;  t.wy = y - (float)y0;  t.wz = z - (float)z0;
  int zy00 = (z0 * 64 + y0) * 64, zy01 = (z0 * 64 + y1) * 64;
  int zy10 = (z1 * 64 + y0) * 64, zy11 = (z1 * 64 + y1) * 64;
  t.i000 = zy00 + x0; t.i001 = zy00 + x1;
  t.i010 = zy01 + x0; t.i011 = zy01 + x1;
  t.i100 = zy10 + x0; t.i101 = zy10 + x1;
  t.i110 = zy11 + x0; t.i111 = zy11 + x1;
  return t;
}

__device__ inline float tri_gather(const float* __restrict__ v, const Tri& t) {
  float c00 = v[t.i000] + t.wx * (v[t.i001] - v[t.i000]);
  float c01 = v[t.i010] + t.wx * (v[t.i011] - v[t.i010]);
  float c10 = v[t.i100] + t.wx * (v[t.i101] - v[t.i100]);
  float c11 = v[t.i110] + t.wx * (v[t.i111] - v[t.i110]);
  float c0 = c00 + t.wy * (c01 - c00);
  float c1 = c10 + t.wy * (c11 - c10);
  return c0 + t.wz * (c1 - c0);
}

// ---------------------------------------------------------------------------
// K_pre: fold the linear chain.  T = W2*W1 [64x67], Tc = Wc2*Wc1 [64x67];
// M_k = Wsum_k * T.  Emit bf16 B-matrices (B[j][o] = M_k[o][j]), f32 coord
// weights, and the fully folded bias.
// ---------------------------------------------------------------------------
__global__ __launch_bounds__(256) void k_pre(
    const float* __restrict__ w_diff1, const float* __restrict__ b_diff1,
    const float* __restrict__ w_diff2, const float* __restrict__ b_diff2,
    const float* __restrict__ w_cen1,  const float* __restrict__ b_cen1,
    const float* __restrict__ w_cen2,  const float* __restrict__ b_cen2,
    const float* __restrict__ w_sum,   const float* __restrict__ b_sum,
    unsigned short* __restrict__ wB,   // [27][64(j)][64(o)] bf16
    float* __restrict__ coordW,        // [26][3][64] f32
    float* __restrict__ cenCoordW,     // [3][64] f32
    float* __restrict__ biasAll)       // [64] f32
{
  __shared__ float sT[64 * 67];
  __shared__ float sTc[64 * 67];
  __shared__ float su[64];
  __shared__ float suc[64];
  const int tid = threadIdx.x;

  for (int e = tid; e < 64 * 67; e += 256) {
    int c = e / 67, j = e % 67;
    float s = 0.f, sc = 0.f;
    for (int m = 0; m < 64; ++m) {
      s  += w_diff2[c * 64 + m] * w_diff1[m * 67 + j];
      sc += w_cen2 [c * 64 + m] * w_cen1 [m * 67 + j];
    }
    sT[e] = s;  sTc[e] = sc;
  }
  for (int c = tid; c < 64; c += 256) {
    float s = b_diff2[c], sc = b_cen2[c];
    for (int m = 0; m < 64; ++m) {
      s  += w_diff2[c * 64 + m] * b_diff1[m];
      sc += w_cen2 [c * 64 + m] * b_cen1 [m];
    }
    su[c] = s;  suc[c] = sc;
  }
  __syncthreads();

  // Feature B-matrices, slots 0..25  (k = kk+1)
  for (int e = tid; e < 26 * 64 * 64; e += 256) {
    int kk = e >> 12, rem = e & 4095, j = rem >> 6, o = rem & 63, k = kk + 1;
    float s = 0.f;
    for (int c = 0; c < 64; ++c)
      s += w_sum[(o * 64 + c) * 27 + k] * sT[c * 67 + j];
    wB[e] = f2bf(s);
  }
  // Slot 26: center branch (abs feats13 path)
  for (int e = tid; e < 4096; e += 256) {
    int j = e >> 6, o = e & 63;
    wB[26 * 4096 + e] = f2bf(sTc[o * 67 + j]);
  }
  // Coordinate weights (channels 64..66 of the effective matrices)
  for (int e = tid; e < 26 * 3 * 64; e += 256) {
    int kk = e / 192, j = (e / 64) % 3, o = e & 63, k = kk + 1;
    float s = 0.f;
    for (int c = 0; c < 64; ++c)
      s += w_sum[(o * 64 + c) * 27 + k] * sT[c * 67 + 64 + j];
    coordW[e] = s;
  }
  for (int e = tid; e < 192; e += 256) {
    int j = e >> 6, o = e & 63;
    cenCoordW[e] = sTc[o * 67 + 64 + j];
  }
  // Folded bias: b_sum + sum_k Wsum_k*(W2 b1 + b2) + (Wc2 bc1 + bc2)
  for (int o = tid; o < 64; o += 256) {
    float s = b_sum[o] + suc[o];
    for (int c = 0; c < 64; ++c) {
      float rs = 0.f;
      for (int k = 0; k < 27; ++k) rs += w_sum[(o * 64 + c) * 27 + k];
      s += rs * su[c];
    }
    biasAll[o] = s;
  }
}

// ---------------------------------------------------------------------------
// K_A: sample feats at vertices, compute learned 27x3 offsets -> neigh coords
// ---------------------------------------------------------------------------
__global__ __launch_bounds__(128) void k_delta(
    const float* __restrict__ vox, const float* __restrict__ verts,
    const float* __restrict__ w_delta, const float* __restrict__ b_delta,
    float* __restrict__ neigh)          // [NPTS][27][3]
{
  __shared__ float sF[128 * 65];
  const int tid = threadIdx.x;
  const int p = blockIdx.x * 128 + tid;
  const int b = p >> 15;
  const float vx = verts[p * 3 + 0], vy = verts[p * 3 + 1], vz = verts[p * 3 + 2];
  Tri t = tri_setup(vx, vy, vz);
  const float* vol = vox + (size_t)b * 64 * DHW;
  float* my = sF + tid * 65;                 // stride 65 -> conflict-free
  for (int c = 0; c < 64; ++c)
    my[c] = tri_gather(vol + (size_t)c * DHW, t);

  float vc[3] = {vx, vy, vz};
  float* np = neigh + (size_t)p * 81;
  np[0] = vx; np[1] = vy; np[2] = vz;        // neighbor 0 pinned (delta = 0)
  for (int o = 3; o < 81; ++o) {
    float acc = b_delta[o];
    const float* wr = w_delta + o * 64;
    for (int c = 0; c < 64; ++c) acc += wr[c] * my[c];
    np[o] = vc[o % 3] + acc;
  }
}

// ---------------------------------------------------------------------------
// K_P: f32 "partial" = folded bias + coordinate-channel contributions
// ---------------------------------------------------------------------------
__global__ __launch_bounds__(256) void k_partial(
    const float* __restrict__ neigh, const float* __restrict__ coordW,
    const float* __restrict__ cenCoordW, const float* __restrict__ biasAll,
    float* __restrict__ partial)        // [NPTS][64]
{
  __shared__ float sN[4 * 81];
  const int tid = threadIdx.x;
  const float* nbase = neigh + (size_t)blockIdx.x * 4 * 81;
  for (int i = tid; i < 324; i += 256) sN[i] = nbase[i];
  __syncthreads();
  const int lp = tid >> 6, o = tid & 63;
  const float* nb = sN + lp * 81;
  float acc = biasAll[o];
  float x0 = nb[0], y0 = nb[1], z0 = nb[2];
  for (int kk = 0; kk < 26; ++kk) {
    int k = kk + 1;
    acc += (nb[k * 3 + 0] - x0) * coordW[(kk * 3 + 0) * 64 + o];
    acc += (nb[k * 3 + 1] - y0) * coordW[(kk * 3 + 1) * 64 + o];
    acc += (nb[k * 3 + 2] - z0) * coordW[(kk * 3 + 2) * 64 + o];
  }
  acc += nb[39] * cenCoordW[o];
  acc += nb[40] * cenCoordW[64 + o];
  acc += nb[41] * cenCoordW[128 + o];
  partial[(size_t)(blockIdx.x * 4 + lp) * 64 + o] = acc;
}

// ---------------------------------------------------------------------------
// K_B: gather 27 neighbors, store bf16 diffs (k=1..26) + abs feats13 (slot 26)
// ---------------------------------------------------------------------------
__global__ __launch_bounds__(64) void k_gather(
    const float* __restrict__ vox, const float* __restrict__ neigh,
    unsigned short* __restrict__ diffF)   // [B][27][NVERT][64] bf16
{
  const int p = blockIdx.x;
  const int c = threadIdx.x;
  const int b = p >> 15;
  const int n = p & (NVERT - 1);
  const float* vol = vox + (size_t)b * 64 * DHW + (size_t)c * DHW;
  const float* nb = neigh + (size_t)p * 81;
  float f0 = 0.f;
  for (int k = 0; k < 27; ++k) {
    Tri t = tri_setup(nb[k * 3 + 0], nb[k * 3 + 1], nb[k * 3 + 2]);
    float f = tri_gather(vol, t);
    if (k == 0) {
      f0 = f;
    } else {
      diffF[(((size_t)b * 27 + (k - 1)) * NVERT + n) * 64 + c] = f2bf(f - f0);
    }
    if (k == 13)
      diffF[(((size_t)b * 27 + 26) * NVERT + n) * 64 + c] = f2bf(f);
  }
}

// ---------------------------------------------------------------------------
// K_C: WMMA GEMM.  One wave = 16 points x 64 outputs; 27 slots x K=64
// -> 216 v_wmma_f32_16x16x32_bf16 per wave.  All 27 weight tiles (216 KB)
// are DMA'd into LDS once at kernel start via TENSOR_LOAD_TO_LDS (TDM),
// so the main loop is barrier-free: global A loads + LDS B loads + WMMA.
// ---------------------------------------------------------------------------
__global__ __launch_bounds__(256) void k_wmma(
    const unsigned short* __restrict__ diffF,
    const unsigned short* __restrict__ wB,
    const float* __restrict__ partial,
    float* __restrict__ out)
{
  extern __shared__ unsigned short sB[];    // 27 * 4096 bf16 = 216 KB dynamic
  const int tid  = threadIdx.x;
  const int lane = tid & 31;
  const int wv   = tid >> 5;
  const int tile = blockIdx.x * 8 + wv;     // 4096 tiles total
  const int p0   = tile << 4;
  const int b    = p0 >> 15;
  const int n0   = p0 & (NVERT - 1);
  const int mrow = lane & 15;
  const int hi   = lane >> 4;
  const int khalf = hi << 3;                // A-frag K base: 0 or 8

  // --- One-shot TDM preload of the folded weight tensor into LDS ---------
  // D# (2D): 27 lines x 4096 bf16, line stride 4096 elems, data_size=2B.
  if (wv == 0) {
    unsigned long long ga = (unsigned long long)(size_t)wB;   // tile start
    unsigned int ldsOff = __builtin_amdgcn_groupstaticsize(); // dynamic base
    u32x4 g0 = { 1u,                                  // count=1, valid D#
                 ldsOff,                               // lds_addr
                 (unsigned int)(ga & 0xFFFFFFFFu),     // global_addr[31:0]
                 (unsigned int)((ga >> 32) & 0x01FFFFFFu) | 0x80000000u }; // [56:32] | type=2
    i32x8 g1 = { (int)0x00010000,   // data_size=1 (2 bytes), no mask/pad
                 (int)0x10000000,   // tensor_dim0 = 4096 (low16 in [31:16])
                 (int)0x001B0000,   // tensor_dim1 = 27  (low16 in [31:16])
                 (int)0x10000000,   // tile_dim0 = 4096  ([31:16])
                 27,                // tile_dim1 = 27, tile_dim2 = 0
                 4096,              // tensor_dim0_stride[31:0]
                 0, 0 };            // stride hi / dim1_stride
    i32x4 g2 = { 0, 0, 0, 0 };
    i32x4 g3 = { 0, 0, 0, 0 };
    i32x8 g4 = { 0, 0, 0, 0, 0, 0, 0, 0 }; // extra group (clang-23 6-arg form)
    __builtin_amdgcn_tensor_load_to_lds(g0, g1, g2, g3, g4, 0);
    __builtin_amdgcn_s_wait_tensorcnt(0);
  }
  __syncthreads();

  v8f acc[4];
  #pragma unroll
  for (int nt = 0; nt < 4; ++nt) {
    #pragma unroll
    for (int r = 0; r < 8; ++r) {
      int prow = p0 + r + (hi << 3);        // C layout: M = r + 8*hi, N = mrow
      acc[nt][r] = partial[(size_t)prow * 64 + nt * 16 + mrow];
    }
  }

  for (int kk = 0; kk < 27; ++kk) {
    union { uint4 q[2]; v16bf v; } afrag[2], bfrag;
    const unsigned short* arow =
        diffF + (((size_t)b * 27 + kk) * NVERT + (n0 + mrow)) * 64 + khalf;
    #pragma unroll
    for (int ch = 0; ch < 2; ++ch) {        // K chunks of 32
      afrag[ch].q[0] = *(const uint4*)(arow + ch * 32);       // K kb+0..7
      afrag[ch].q[1] = *(const uint4*)(arow + ch * 32 + 16);  // K kb+16..23
    }
    const unsigned short* bt = sB + kk * 4096;
    #pragma unroll
    for (int ch = 0; ch < 2; ++ch) {
      const int krow = ch * 32 + khalf;     // B layout: lane L holds column N=L%16, K in table order
      #pragma unroll
      for (int nt = 0; nt < 4; ++nt) {
        const unsigned short* bp = bt + krow * 64 + nt * 16 + mrow;
        unsigned short* bs = (unsigned short*)&bfrag;
        #pragma unroll
        for (int s = 0; s < 8; ++s) {
          bs[s] = bp[s * 64];
          bs[8 + s] = bp[(16 + s) * 64];
        }
        acc[nt] = __builtin_amdgcn_wmma_f32_16x16x32_bf16(
            false, afrag[ch].v, false, bfrag.v,
            (short)0, acc[nt], false, false);
      }
    }
  }

  #pragma unroll
  for (int nt = 0; nt < 4; ++nt) {
    #pragma unroll
    for (int r = 0; r < 8; ++r) {
      int prow = p0 + r + (hi << 3);
      out[(size_t)prow * 64 + nt * 16 + mrow] = acc[nt][r];
    }
  }
}

// ---------------------------------------------------------------------------
extern "C" void kernel_launch(void* const* d_in, const int* in_sizes, int n_in,
                              void* d_out, int out_size, void* d_ws, size_t ws_size,
                              hipStream_t stream) {
  (void)in_sizes; (void)n_in; (void)out_size; (void)ws_size;

  const float* vox     = (const float*)d_in[0];
  const float* verts   = (const float*)d_in[1];
  const float* w_delta = (const float*)d_in[2];
  const float* b_delta = (const float*)d_in[3];
  const float* w_diff1 = (const float*)d_in[4];
  const float* b_diff1 = (const float*)d_in[5];
  const float* w_diff2 = (const float*)d_in[6];
  const float* b_diff2 = (const float*)d_in[7];
  const float* w_cen1  = (const float*)d_in[8];
  const float* b_cen1  = (const float*)d_in[9];
  const float* w_cen2  = (const float*)d_in[10];
  const float* b_cen2  = (const float*)d_in[11];
  const float* w_sum   = (const float*)d_in[12];
  const float* b_sum   = (const float*)d_in[13];
  float* out = (float*)d_out;

  // Workspace carving (all offsets multiples of 256 B); total ~253 MiB.
  char* w = (char*)d_ws;
  float* neigh = (float*)w;                  w += (size_t)NPTS * 81 * 4;      // 21,233,664
  float* partial = (float*)w;                w += (size_t)NPTS * 64 * 4;      // 16,777,216
  unsigned short* diffF = (unsigned short*)w; w += (size_t)NB * 27 * NVERT * 64 * 2; // 226,492,416
  unsigned short* wB = (unsigned short*)w;   w += (size_t)27 * 64 * 64 * 2;   // 221,184
  float* coordW = (float*)w;                 w += (size_t)26 * 3 * 64 * 4;    // 19,968
  float* cenCoordW = (float*)w;              w += (size_t)3 * 64 * 4;         // 768
  float* biasAll = (float*)w;                w += (size_t)64 * 4;             // 256

  k_pre<<<1, 256, 0, stream>>>(w_diff1, b_diff1, w_diff2, b_diff2,
                               w_cen1, b_cen1, w_cen2, b_cen2,
                               w_sum, b_sum, wB, coordW, cenCoordW, biasAll);
  k_delta<<<NPTS / 128, 128, 0, stream>>>(vox, verts, w_delta, b_delta, neigh);
  k_partial<<<NPTS / 4, 256, 0, stream>>>(neigh, coordW, cenCoordW, biasAll, partial);
  k_gather<<<NPTS, 64, 0, stream>>>(vox, neigh, diffF);
  k_wmma<<<512, 256, (size_t)27 * 4096 * 2, stream>>>(diffF, wB, partial, out);
}
